// CustomMoERouter_54494545052069
// MI455X (gfx1250) — compile-verified, matrix operand order checked by reading, not run on hardware
//
#include <hip/hip_runtime.h>
#include <hip/hip_bf16.h>

typedef __attribute__((ext_vector_type(16))) __bf16          v16bf;
typedef __attribute__((ext_vector_type(8)))  float           v8f;
typedef __attribute__((ext_vector_type(16))) unsigned short  v16u;
typedef __attribute__((ext_vector_type(8)))  unsigned short  v8u;
typedef __attribute__((ext_vector_type(8)))  unsigned int    v8ui;
typedef __attribute__((ext_vector_type(4)))  unsigned short  v4u;
typedef __attribute__((ext_vector_type(4)))  float           v4f;

#define T_TOK 16384
#define H_DIM 4096
#define E_DIM 64
#define TOPK  8
#define KSPLIT 4   // K-quarters per tile; 4 waves/block, one per quarter

// Pack two f32 into two bf16 (round-to-nearest, ties away) in ONE v_perm_b32.
__device__ __forceinline__ unsigned pack2_bf16(float lo, float hi) {
  unsigned a = __builtin_bit_cast(unsigned, lo) + 0x8000u;
  unsigned b = __builtin_bit_cast(unsigned, hi) + 0x8000u;
  return __builtin_amdgcn_perm(b, a, 0x07060302u);
}

__device__ __forceinline__ unsigned short f2bf_rne(float f) {
  unsigned u = __builtin_bit_cast(unsigned, f);
  unsigned r = u + 0x7FFFu + ((u >> 16) & 1u);
  return (unsigned short)(r >> 16);
}

__device__ __forceinline__ float sigmoidf_fast(float x) {
  return 1.0f / (1.0f + __expf(-x));
}

// ---------------------------------------------------------------------------
// Kernel 0: convert W [64,4096] f32 -> bf16 in workspace (512 KB, L2-resident)
// ---------------------------------------------------------------------------
__global__ void wconv_kernel(const float* __restrict__ W,
                             unsigned short* __restrict__ Wbf) {
  int i = (blockIdx.x * blockDim.x + threadIdx.x) * 4;
  v4f w = *(const v4f*)(W + i);
  v4u o;
  o[0] = f2bf_rne(w[0]); o[1] = f2bf_rne(w[1]);
  o[2] = f2bf_rne(w[2]); o[3] = f2bf_rne(w[3]);
  *(v4u*)(Wbf + i) = o;
}

// 32x16 (KxN) bf16 B fragment: lane = n, elems 0..7 at k+koff, 8..15 at +16.
__device__ __forceinline__ v16bf load_bfrag(const unsigned short* p) {
  v8u lo = *(const v8u*)p;
  v8u hi = *(const v8u*)(p + 16);
  v16u u;
#pragma unroll
  for (int i = 0; i < 8; ++i) { u[i] = lo[i]; u[8 + i] = hi[i]; }
  return __builtin_bit_cast(v16bf, u);
}

// ---------------------------------------------------------------------------
// Kernel 1: router GEMM, K split across 4 waves of a 128-thread block.
// Block = one 16-row M tile; wave q accumulates K in [q*1024, q*1024+1024).
// Deterministic fixed-order combine through LDS, then bias+sigmoid+store.
// ---------------------------------------------------------------------------
__global__ void __launch_bounds__(128, 1)
router_gemm_kernel(const float* __restrict__ hs,
                   const unsigned short* __restrict__ Wbf,
                   const float* __restrict__ bias,
                   float* __restrict__ probs) {
  __shared__ float red[KSPLIT][16][E_DIM];   // 16 KB

  const int lane = threadIdx.x & 31;
  const int wave = threadIdx.x >> 5;          // K-quarter id, 0..3
  const int row0 = blockIdx.x * 16;
  const int m    = lane & 15;
  const int koff = (lane >> 4) * 8;
  const int k0   = wave * (H_DIM / KSPLIT);   // 1024 per quarter

  const float*          aptr = hs  + (size_t)(row0 + m) * H_DIM + k0 + koff;
  const unsigned short* bptr = Wbf + (size_t)m * H_DIM + k0 + koff;

  v8f acc0 = {}, acc1 = {}, acc2 = {}, acc3 = {};

#pragma unroll 4
  for (int k = 0; k < H_DIM / KSPLIT; k += 32) {   // 32 iterations
    // A fragment: 16 f32 non-temporal (single-use 256MB stream) -> bf16
    v4f a0 = __builtin_nontemporal_load((const v4f*)(aptr));
    v4f a1 = __builtin_nontemporal_load((const v4f*)(aptr + 4));
    v4f a2 = __builtin_nontemporal_load((const v4f*)(aptr + 16));
    v4f a3 = __builtin_nontemporal_load((const v4f*)(aptr + 20));
    v8ui au;
    au[0] = pack2_bf16(a0[0], a0[1]);
    au[1] = pack2_bf16(a0[2], a0[3]);
    au[2] = pack2_bf16(a1[0], a1[1]);
    au[3] = pack2_bf16(a1[2], a1[3]);
    au[4] = pack2_bf16(a2[0], a2[1]);
    au[5] = pack2_bf16(a2[2], a2[3]);
    au[6] = pack2_bf16(a3[0], a3[1]);
    au[7] = pack2_bf16(a3[2], a3[3]);
    v16bf afrag = __builtin_bit_cast(v16bf, au);

    // B fragments for the 4 expert tiles (bf16, L2-resident)
    v16bf b0 = load_bfrag(bptr);
    v16bf b1 = load_bfrag(bptr + (size_t)16 * H_DIM);
    v16bf b2 = load_bfrag(bptr + (size_t)32 * H_DIM);
    v16bf b3 = load_bfrag(bptr + (size_t)48 * H_DIM);

    acc0 = __builtin_amdgcn_wmma_f32_16x16x32_bf16(false, afrag, false, b0,
                                                   (short)0, acc0, false, false);
    acc1 = __builtin_amdgcn_wmma_f32_16x16x32_bf16(false, afrag, false, b1,
                                                   (short)0, acc1, false, false);
    acc2 = __builtin_amdgcn_wmma_f32_16x16x32_bf16(false, afrag, false, b2,
                                                   (short)0, acc2, false, false);
    acc3 = __builtin_amdgcn_wmma_f32_16x16x32_bf16(false, afrag, false, b3,
                                                   (short)0, acc3, false, false);

    aptr += 32;
    bptr += 32;
  }

  // ---- Stage partials to LDS (C/D layout: lanes 0-15 -> M=i, 16-31 -> M=8+i)
#pragma unroll
  for (int i = 0; i < 8; ++i) {
    int rm = (lane < 16) ? i : (8 + i);
    red[wave][rm][m]      = acc0[i];
    red[wave][rm][m + 16] = acc1[i];
    red[wave][rm][m + 32] = acc2[i];
    red[wave][rm][m + 48] = acc3[i];
  }
  __syncthreads();

  // ---- Fixed-order reduce + bias + sigmoid. 128 threads x 8 elems = 16x64.
  const int tid = threadIdx.x;
  const int e   = tid & 63;              // loop-invariant expert id
  const int rb  = tid >> 6;              // 0 or 1
  const float be = bias[e];
#pragma unroll
  for (int j = 0; j < 8; ++j) {
    int r = j * 2 + rb;                  // rows 0..15
    float s = ((red[0][r][e] + red[1][r][e]) +
               (red[2][r][e] + red[3][r][e])) + be;
    probs[(size_t)(row0 + r) * E_DIM + e] = sigmoidf_fast(s);
  }
}

// ---------------------------------------------------------------------------
// Kernel 2: per-row top-8 (JAX tie-break: lower index wins on ties),
// normalize, write weights + int32 indices.
// ---------------------------------------------------------------------------
__global__ void topk_kernel(const float* __restrict__ probs,
                            float* __restrict__ wout,
                            int* __restrict__ iout) {
  int row = blockIdx.x * blockDim.x + threadIdx.x;
  if (row >= T_TOK) return;
  const float* pr = probs + (size_t)row * E_DIM;

  float v[TOPK];
  int   id[TOPK];
#pragma unroll
  for (int j = 0; j < TOPK; ++j) { v[j] = -3.0e38f; id[j] = 0; }

  for (int e = 0; e < E_DIM; ++e) {
    float p = pr[e];
    if (p > v[TOPK - 1]) {
      int j = TOPK - 1;
#pragma unroll
      for (int s = TOPK - 1; s > 0; --s) {
        if (p > v[s - 1]) { v[s] = v[s - 1]; id[s] = id[s - 1]; j = s - 1; }
      }
      v[j] = p; id[j] = e;
    }
  }

  float sum = 0.0f;
#pragma unroll
  for (int j = 0; j < TOPK; ++j) sum += v[j];
  float inv = 1.0f / sum;

#pragma unroll
  for (int j = 0; j < TOPK; ++j) {
    wout[(size_t)row * TOPK + j] = v[j] * inv;
    iout[(size_t)row * TOPK + j] = id[j];
  }
}

// ---------------------------------------------------------------------------
extern "C" void kernel_launch(void* const* d_in, const int* in_sizes, int n_in,
                              void* d_out, int out_size, void* d_ws, size_t ws_size,
                              hipStream_t stream) {
  const float* hs = (const float*)d_in[0];   // [16384, 4096]
  const float* W  = (const float*)d_in[1];   // [64, 4096]
  const float* b  = (const float*)d_in[2];   // [64]

  // d_out: weights [T*8] f32 | indices [T*8] i32 | probs [T*64] f32
  float* wout  = (float*)d_out;
  int*   iout  = (int*)((float*)d_out + T_TOK * TOPK);
  float* probs = (float*)d_out + 2 * T_TOK * TOPK;

  unsigned short* Wbf = (unsigned short*)d_ws;  // 64*4096*2 = 512 KB

  // 1) W -> bf16 (deterministic per launch)
  wconv_kernel<<<(E_DIM * H_DIM) / (256 * 4), 256, 0, stream>>>(W, Wbf);

  // 2) GEMM + reduce + sigmoid: 1024 blocks x 4 waves (one per K-quarter)
  router_gemm_kernel<<<T_TOK / 16, 128, 0, stream>>>(hs, Wbf, b, probs);

  // 3) top-8 + normalize
  topk_kernel<<<T_TOK / 256, 256, 0, stream>>>(probs, wout, iout);
}
